// GraphDecoder_36326833389700
// MI455X (gfx1250) — matrix operand order repaired and checked by reading, not compiled, not run on hardware
//
#include <hip/hip_runtime.h>

// ---------------------------------------------------------------------------
// GraphSAGE 4-layer GNN for MI455X (gfx1250), wave32 + WMMA bf16 matrix cores.
// Per layer:  y = relu(LN(mean_agg @ w_msg + x @ w_root + b))
//
// Design (roofline-driven):
//  * All fp32->bf16 hi/lo error-compensation splits happen in bandwidth-bound
//    prep passes (weights pre-swizzled into WMMA B-fragment lane order;
//    features stored as bf16 hi/lo planes).  The GEMM inner loop is pure
//    contiguous vector loads + v_wmma_f32_16x16x32_bf16 (3 per K-tile:
//    ah*bh + ah*bl + al*bh, f32 accumulate ~ fp32-accurate).
//  * Edge aggregation gathers bf16 hi+lo (8B hi + 8B lo per 4 feats, same as 16B f32),
//    reconstructs f32, and atomic-adds; the <=102MB feature planes stay
//    resident in the 192MB L2, so gathers/atomics never touch HBM.
//  * LayerNorm+ReLU fused into the GEMM epilogue via LDS + wave32 shuffles.
// ---------------------------------------------------------------------------

typedef __bf16 bf16_t;
typedef __attribute__((ext_vector_type(4)))  __bf16 v4bf;
typedef __attribute__((ext_vector_type(8)))  __bf16 v8bf;
typedef __attribute__((ext_vector_type(16))) __bf16 v16bf;
typedef __attribute__((ext_vector_type(8)))  float  v8f;

// ----------------------------- degree counting -----------------------------
__global__ void count_deg_kernel(const int* __restrict__ dst, float* __restrict__ counts, int E) {
    for (int e = blockIdx.x * blockDim.x + threadIdx.x; e < E; e += gridDim.x * blockDim.x)
        atomicAdd(counts + dst[e], 1.0f);
}

__global__ void inv_counts_kernel(float* __restrict__ c, int n) {
    int i = blockIdx.x * blockDim.x + threadIdx.x;
    if (i < n) c[i] = 1.0f / fmaxf(c[i], 1.0f);
}

// ------------- weights -> bf16 hi/lo in WMMA B-fragment lane order ---------
// Fragment layout consumed by the GEMM:  for K-tile kt, col-tile nt,
//   frag[(((kt*NT + nt)*32 + lane)*16 + idx)] = W[kt*32 + (idx&7) + 8*(lane>>4) + 16*(idx>>3)]
//                                                [nt*16 + (lane&15)]
// Zero-padded when doutt > dout (layer 4: 9 -> 16).
__global__ void split_w_frag_kernel(const float* __restrict__ w, bf16_t* __restrict__ hi,
                                    bf16_t* __restrict__ lo, int din, int dout, int doutt) {
    const int ntiles = doutt >> 4;
    const int total = din * doutt;
    for (int t = blockIdx.x * blockDim.x + threadIdx.x; t < total; t += gridDim.x * blockDim.x) {
        const int idx  = t & 15;
        const int lane = (t >> 4) & 31;
        const int f    = t >> 9;
        const int nt   = f % ntiles;
        const int kt   = f / ntiles;
        const int k = kt * 32 + (idx & 7) + 8 * (lane >> 4) + 16 * (idx >> 3);
        const int n = nt * 16 + (lane & 15);
        const float v = (n < dout) ? w[k * dout + n] : 0.0f;
        const bf16_t h = (bf16_t)v;
        hi[t] = h;
        lo[t] = (bf16_t)(v - (float)h);
    }
}

// --------- f32 plane (optionally row-scaled) -> bf16 hi/lo planes ----------
__global__ void split_plane_kernel(const float* __restrict__ in, const float* __restrict__ rowscale,
                                   bf16_t* __restrict__ hi, bf16_t* __restrict__ lo,
                                   long n, int log2d) {
    const long stride = (long)gridDim.x * blockDim.x;
    for (long i = (long)blockIdx.x * blockDim.x + threadIdx.x; i < n; i += stride) {
        float v = in[i];
        if (rowscale) v *= rowscale[i >> log2d];
        const bf16_t h = (bf16_t)v;
        hi[i] = h;
        lo[i] = (bf16_t)(v - (float)h);
    }
}

// --------------------- edge-parallel neighbor sum (L2 atomics) -------------
// f32-source variant (layer 1 input z).
__global__ void agg_f32_kernel(const float* __restrict__ x, const int* __restrict__ src,
                               const int* __restrict__ dst, float* __restrict__ agg,
                               int E, int d, int log2chunks) {
    const int chunks = d >> 2;
    const long total = (long)E * chunks;
    const long stride = (long)gridDim.x * blockDim.x;
    for (long i = (long)blockIdx.x * blockDim.x + threadIdx.x; i < total; i += stride) {
        const int e  = (int)(i >> log2chunks);
        const int c4 = ((int)i & (chunks - 1)) << 2;
        const int s = src[e];
        const int t = dst[e];
        const float4 v = *reinterpret_cast<const float4*>(x + (long)s * d + c4);
        float* o = agg + (long)t * d + c4;
        atomicAdd(o + 0, v.x);
        atomicAdd(o + 1, v.y);
        atomicAdd(o + 2, v.z);
        atomicAdd(o + 3, v.w);
    }
}

// bf16 hi/lo source variant (layers 2-4): same gather bytes as f32 (8B hi + 8B lo per 4 feats).
__global__ void agg_bf16_kernel(const bf16_t* __restrict__ xh, const bf16_t* __restrict__ xl,
                                const int* __restrict__ src, const int* __restrict__ dst,
                                float* __restrict__ agg, int E, int d, int log2chunks) {
    const int chunks = d >> 2;
    const long total = (long)E * chunks;
    const long stride = (long)gridDim.x * blockDim.x;
    for (long i = (long)blockIdx.x * blockDim.x + threadIdx.x; i < total; i += stride) {
        const int e  = (int)(i >> log2chunks);
        const int c4 = ((int)i & (chunks - 1)) << 2;
        const int s = src[e];
        const int t = dst[e];
        const v4bf h = *reinterpret_cast<const v4bf*>(xh + (long)s * d + c4);
        const v4bf l = *reinterpret_cast<const v4bf*>(xl + (long)s * d + c4);
        float* o = agg + (long)t * d + c4;
        atomicAdd(o + 0, (float)h[0] + (float)l[0]);
        atomicAdd(o + 1, (float)h[1] + (float)l[1]);
        atomicAdd(o + 2, (float)h[2] + (float)l[2]);
        atomicAdd(o + 3, (float)h[3] + (float)l[3]);
    }
}

// ---------------- fused dual-GEMM + bias + LayerNorm + ReLU ----------------
// Block covers 16 node rows x DOUTT cols; one 16x16 WMMA C tile per wave.
// K-loop runs over [mean | x] (both pre-split bf16 hi/lo planes); B comes
// from pre-swizzled fragments (one contiguous 32B load per fragment).
template <int DIN, int DOUT, int DOUTT, bool LN>
__global__ __launch_bounds__(DOUTT * 2) void sage_gemm_ln_kernel(
    const bf16_t* __restrict__ xh, const bf16_t* __restrict__ xl,   // root planes  N x DIN
    const bf16_t* __restrict__ mh, const bf16_t* __restrict__ ml,   // mean planes  N x DIN
    const bf16_t* __restrict__ fmh, const bf16_t* __restrict__ fml, // w_msg  frags
    const bf16_t* __restrict__ frh, const bf16_t* __restrict__ frl, // w_root frags
    const float* __restrict__ bias,
    const float* __restrict__ gamma,
    const float* __restrict__ beta,
    bf16_t* __restrict__ yh, bf16_t* __restrict__ yl,               // LN outputs (hi/lo)
    float* __restrict__ yf)                                         // non-LN output (f32)
{
    constexpr int NT = DOUTT / 16;   // col tiles == waves
    constexpr int KT = DIN / 32;     // K tiles per phase
    const int lane = threadIdx.x & 31;
    const int wv   = threadIdx.x >> 5;
    const int half = lane >> 4;
    const int l15  = lane & 15;
    const int m0   = blockIdx.x * 16;
    const int arow = m0 + l15;          // A row owned by this lane
    const int col  = wv * 16 + l15;     // B/C column owned by this lane
    const size_t lane_frag = ((size_t)wv * 32 + lane) * 16;

    v8f c = {};

    #pragma unroll
    for (int phase = 0; phase < 2; ++phase) {
        const bf16_t* ah_p = (phase ? xh : mh) + (size_t)arow * DIN;
        const bf16_t* al_p = (phase ? xl : ml) + (size_t)arow * DIN;
        const bf16_t* bh_p = (phase ? frh : fmh) + lane_frag;
        const bf16_t* bl_p = (phase ? frl : fml) + lane_frag;

        for (int kt = 0; kt < KT; ++kt) {
            // A 16x32 bf16 layout: per lane, two runs of 8 consecutive K
            const int base = kt * 32 + 8 * half;
            const v8bf a0 = *(const v8bf*)(ah_p + base);
            const v8bf a1 = *(const v8bf*)(ah_p + base + 16);
            const v8bf e0 = *(const v8bf*)(al_p + base);
            const v8bf e1 = *(const v8bf*)(al_p + base + 16);
            const v16bf ah = __builtin_shufflevector(a0, a1, 0, 1, 2, 3, 4, 5, 6, 7,
                                                     8, 9, 10, 11, 12, 13, 14, 15);
            const v16bf al = __builtin_shufflevector(e0, e1, 0, 1, 2, 3, 4, 5, 6, 7,
                                                     8, 9, 10, 11, 12, 13, 14, 15);
            // B fragment: contiguous 32B per lane, pre-swizzled
            const size_t foff = (size_t)kt * (NT * 32 * 16);
            const v16bf bh = *(const v16bf*)(bh_p + foff);
            const v16bf bl = *(const v16bf*)(bl_p + foff);
            if (kt + 1 < KT)  // gfx1250 global_prefetch_b8 on next K-tile frag
                __builtin_prefetch(bh_p + foff + (size_t)NT * 32 * 16, 0, 1);

            c = __builtin_amdgcn_wmma_f32_16x16x32_bf16(false, ah, false, bh, (short)0, c, false, false);
            c = __builtin_amdgcn_wmma_f32_16x16x32_bf16(false, ah, false, bl, (short)0, c, false, false);
            c = __builtin_amdgcn_wmma_f32_16x16x32_bf16(false, al, false, bh, (short)0, c, false, false);
        }
    }

    if constexpr (LN) {
        __shared__ float tile[16][DOUTT + 4];
        const float bcol = bias[col];
        // C layout: VGPR r -> row r (lanes 0-15) / r+8 (lanes 16-31)
        #pragma unroll
        for (int r = 0; r < 8; ++r)
            tile[r + 8 * half][col] = c[r] + bcol;
        __syncthreads();

        for (int r = wv; r < 16; r += NT) {
            float s = 0.0f, s2 = 0.0f;
            #pragma unroll
            for (int ci = lane; ci < DOUT; ci += 32) {
                const float t = tile[r][ci];
                s += t;
                s2 += t * t;
            }
            #pragma unroll
            for (int off = 16; off > 0; off >>= 1) {
                s  += __shfl_xor(s,  off, 32);
                s2 += __shfl_xor(s2, off, 32);
            }
            const float mean = s * (1.0f / DOUT);
            const float var  = s2 * (1.0f / DOUT) - mean * mean;
            const float rstd = rsqrtf(var + 1e-5f);
            const long grow = m0 + r;
            #pragma unroll
            for (int ci = lane; ci < DOUT; ci += 32) {
                float t = (tile[r][ci] - mean) * rstd * gamma[ci] + beta[ci];
                t = fmaxf(t, 0.0f);
                const bf16_t h = (bf16_t)t;          // emit next layer's hi/lo planes
                yh[grow * DOUT + ci] = h;
                yl[grow * DOUT + ci] = (bf16_t)(t - (float)h);
            }
        }
    } else {
        if (col < DOUT) {
            const float bcol = bias[col];
            #pragma unroll
            for (int r = 0; r < 8; ++r) {
                const long grow = m0 + r + 8 * half;
                yf[grow * DOUT + col] = c[r] + bcol;
            }
        }
    }
}

// ---------------------------------------------------------------------------
extern "C" void kernel_launch(void* const* d_in, const int* in_sizes, int n_in,
                              void* d_out, int out_size, void* d_ws, size_t ws_size,
                              hipStream_t stream) {
    const float* z  = (const float*)d_in[0];
    const int*   ei = (const int*)d_in[1];
    const int N = in_sizes[0] / 128;   // 100000
    const int E = in_sizes[1] / 2;     // 1600000
    const int* src = ei;
    const int* dst = ei + E;

    const float* wm[4] = {(const float*)d_in[2],  (const float*)d_in[7],
                          (const float*)d_in[12], (const float*)d_in[17]};
    const float* wr[4] = {(const float*)d_in[3],  (const float*)d_in[8],
                          (const float*)d_in[13], (const float*)d_in[18]};
    const float* bb[4] = {(const float*)d_in[4],  (const float*)d_in[9],
                          (const float*)d_in[14], (const float*)d_in[19]};
    const float* gg[3] = {(const float*)d_in[5],  (const float*)d_in[10],
                          (const float*)d_in[15]};
    const float* be[3] = {(const float*)d_in[6],  (const float*)d_in[11],
                          (const float*)d_in[16]};
    const int din_l[4]   = {128, 256, 256, 128};
    const int dout_l[4]  = {256, 256, 128, 9};
    const int doutt_l[4] = {256, 256, 128, 16};

    // ----- workspace carve (all chunks 32B-aligned by construction) -----
    char* ws = (char*)d_ws;
    float* counts = (float*)ws; ws += (size_t)N * sizeof(float);
    float* agg    = (float*)ws; ws += (size_t)N * 256 * sizeof(float);
    bf16_t* mnh   = (bf16_t*)ws; ws += (size_t)N * 256 * sizeof(bf16_t);
    bf16_t* mnl   = (bf16_t*)ws; ws += (size_t)N * 256 * sizeof(bf16_t);
    bf16_t* zh    = (bf16_t*)ws; ws += (size_t)N * 128 * sizeof(bf16_t);
    bf16_t* zl    = (bf16_t*)ws; ws += (size_t)N * 128 * sizeof(bf16_t);
    bf16_t* x1h   = (bf16_t*)ws; ws += (size_t)N * 256 * sizeof(bf16_t);
    bf16_t* x1l   = (bf16_t*)ws; ws += (size_t)N * 256 * sizeof(bf16_t);
    bf16_t* x2h   = (bf16_t*)ws; ws += (size_t)N * 256 * sizeof(bf16_t);
    bf16_t* x2l   = (bf16_t*)ws; ws += (size_t)N * 256 * sizeof(bf16_t);
    bf16_t* x3h   = x1h;  // x1 planes dead after layer-2 agg+GEMM
    bf16_t* x3l   = x1l;
    bf16_t* fmh[4]; bf16_t* fml[4]; bf16_t* frh[4]; bf16_t* frl[4];
    for (int l = 0; l < 4; ++l) {
        const size_t fsz = (size_t)din_l[l] * doutt_l[l];
        fmh[l] = (bf16_t*)ws; ws += fsz * sizeof(bf16_t);
        fml[l] = (bf16_t*)ws; ws += fsz * sizeof(bf16_t);
        frh[l] = (bf16_t*)ws; ws += fsz * sizeof(bf16_t);
        frl[l] = (bf16_t*)ws; ws += fsz * sizeof(bf16_t);
    }

    // ----- degree counts -> 1/max(deg,1) -----
    hipMemsetAsync(counts, 0, (size_t)N * sizeof(float), stream);
    count_deg_kernel<<<2048, 256, 0, stream>>>(dst, counts, E);
    inv_counts_kernel<<<(N + 255) / 256, 256, 0, stream>>>(counts, N);

    // ----- weights -> pre-swizzled bf16 hi/lo fragments -----
    for (int l = 0; l < 4; ++l) {
        const int total = din_l[l] * doutt_l[l];
        const int blk = (total + 255) / 256;
        split_w_frag_kernel<<<blk, 256, 0, stream>>>(wm[l], fmh[l], fml[l], din_l[l], dout_l[l], doutt_l[l]);
        split_w_frag_kernel<<<blk, 256, 0, stream>>>(wr[l], frh[l], frl[l], din_l[l], dout_l[l], doutt_l[l]);
    }

    // ----- z -> bf16 hi/lo planes -----
    split_plane_kernel<<<4096, 256, 0, stream>>>(z, nullptr, zh, zl, (long)N * 128, 7);

    const int nblk = N / 16;  // 6250

    // ----- layer 1: 128 -> 256, LN+ReLU -----
    hipMemsetAsync(agg, 0, (size_t)N * 128 * sizeof(float), stream);
    agg_f32_kernel<<<4096, 256, 0, stream>>>(z, src, dst, agg, E, 128, 5);
    split_plane_kernel<<<4096, 256, 0, stream>>>(agg, counts, mnh, mnl, (long)N * 128, 7);
    sage_gemm_ln_kernel<128, 256, 256, true><<<nblk, 512, 0, stream>>>(
        zh, zl, mnh, mnl, fmh[0], fml[0], frh[0], frl[0], bb[0], gg[0], be[0],
        x1h, x1l, nullptr);

    // ----- layer 2: 256 -> 256, LN+ReLU -----
    hipMemsetAsync(agg, 0, (size_t)N * 256 * sizeof(float), stream);
    agg_bf16_kernel<<<4096, 256, 0, stream>>>(x1h, x1l, src, dst, agg, E, 256, 6);
    split_plane_kernel<<<4096, 256, 0, stream>>>(agg, counts, mnh, mnl, (long)N * 256, 8);
    sage_gemm_ln_kernel<256, 256, 256, true><<<nblk, 512, 0, stream>>>(
        x1h, x1l, mnh, mnl, fmh[1], fml[1], frh[1], frl[1], bb[1], gg[1], be[1],
        x2h, x2l, nullptr);

    // ----- layer 3: 256 -> 128, LN+ReLU -----
    hipMemsetAsync(agg, 0, (size_t)N * 256 * sizeof(float), stream);
    agg_bf16_kernel<<<4096, 256, 0, stream>>>(x2h, x2l, src, dst, agg, E, 256, 6);
    split_plane_kernel<<<4096, 256, 0, stream>>>(agg, counts, mnh, mnl, (long)N * 256, 8);
    sage_gemm_ln_kernel<256, 128, 128, true><<<nblk, 256, 0, stream>>>(
        x2h, x2l, mnh, mnl, fmh[2], fml[2], frh[2], frl[2], bb[2], gg[2], be[2],
        x3h, x3l, nullptr);

    // ----- layer 4: 128 -> 9, no LN -----
    hipMemsetAsync(agg, 0, (size_t)N * 128 * sizeof(float), stream);
    agg_bf16_kernel<<<4096, 256, 0, stream>>>(x3h, x3l, src, dst, agg, E, 128, 5);
    split_plane_kernel<<<4096, 256, 0, stream>>>(agg, counts, mnh, mnl, (long)N * 128, 7);
    sage_gemm_ln_kernel<128, 9, 16, false><<<nblk, 32, 0, stream>>>(
        x3h, x3l, mnh, mnl, fmh[3], fml[3], frh[3], frl[3], bb[3], nullptr, nullptr,
        nullptr, nullptr, (float*)d_out);
}
